// GRPENetwork_86921548136525
// MI455X (gfx1250) — compile-verified, matrix operand order checked by reading, not run on hardware
//
#include <hip/hip_runtime.h>
#include <stdint.h>

// ---------------- model constants ----------------
#define Lc   6
#define Bc   16
#define Nc   255
#define Dc   768
#define Hc   12
#define DKc  64
#define DFFc 3072
#define Sc   256
#define MAX_HOP   256
#define THOPc     259     // hop table size
#define NETc      25
#define TEDGEc    29
#define TASK_DIST 257
#define UNREACH   258
#define TASK_EDGE 26
#define SELF_EDGE 27
#define NO_EDGE   28
#define SCALEc    0.125f  // 64^-0.5
// combined bias/value table: bins 0..287 = hop (259 used, padded to 288 for
// WMMA K granularity), bins 288..319 = edge (29 used).  Total K = 320.
#define EOFF 288
#define TBP  320

typedef __attribute__((ext_vector_type(16))) _Float16 v16h;
typedef __attribute__((ext_vector_type(8)))  _Float16 h8;
typedef __attribute__((ext_vector_type(8)))  float    v8f;

// ---------------- WMMA helpers (CDNA5 wave32 layouts) ----------------
// A fragment (f16 16x32): lane L holds row (L&15); halves 0..7 = K kb..kb+7,
// halves 8..15 = K kb+16..kb+23, kb = (L>>4)*8.  Two 16B loads per lane.
// B fragment is loaded from a TRANSPOSED tile Bt[N][K] with the same pattern
// (lane -> column N, halves -> K), mirroring the A striping.
__device__ __forceinline__ v16h frag_load(const _Float16* base, int ld) {
  int lane = threadIdx.x & 31;
  const _Float16* p = base + (size_t)(lane & 15) * ld + ((lane >> 4) << 3);
  h8 lo = *reinterpret_cast<const h8*>(p);
  h8 hi = *reinterpret_cast<const h8*>(p + 16);
  v16h r;
#pragma unroll
  for (int i = 0; i < 8; ++i) { r[i] = lo[i]; r[i + 8] = hi[i]; }
  return r;
}

__device__ __forceinline__ v8f wmma32(v16h a, v16h b, v8f c) {
  return __builtin_amdgcn_wmma_f32_16x16x32_f16(false, a, false, b, (short)0, c,
                                                false, false);
}

// ---------------- prep kernels ----------------
__global__ __launch_bounds__(256) void embed_kernel(
    const int* __restrict__ node_x, const float* __restrict__ node_emb,
    const float* __restrict__ task_token, float* __restrict__ xt) {
  int idx = blockIdx.x * 256 + threadIdx.x;
  if (idx >= Bc * Sc * Dc) return;
  int d = idx % Dc;
  int rest = idx / Dc;
  int s = rest % Sc;
  int b = rest / Sc;
  float v;
  if (s == 0) {
    v = task_token[d];
  } else {
    int n = s - 1;
    int e0 = node_x[((size_t)b * Nc + n) * 2 + 0];
    int e1 = node_x[((size_t)b * Nc + n) * 2 + 1];
    v = node_emb[(size_t)e0 * Dc + d] + node_emb[(size_t)e1 * Dc + d];
  }
  xt[idx] = v;
}

__global__ __launch_bounds__(256) void dist_edge_kernel(
    const int* __restrict__ distance, const int* __restrict__ edge_attr,
    int* __restrict__ DT, int* __restrict__ EA) {
  int idx = blockIdx.x * 256 + threadIdx.x;
  if (idx >= Bc * Sc * Sc) return;
  int j = idx & 255;
  int rest = idx >> 8;
  int i = rest & 255;
  int b = rest >> 8;
  int dtv;
  if (i == 0 && j == 0) dtv = 0;
  else if (i == 0 || j == 0) dtv = TASK_DIST;
  else {
    int dd = distance[((size_t)b * Nc + (i - 1)) * Nc + (j - 1)];
    dtv = dd < MAX_HOP ? dd : MAX_HOP;
    if (dtv == -1) dtv = UNREACH;
  }
  int eav = 0;
  if (i > 0 && j > 0) eav = edge_attr[((size_t)b * Nc + (i - 1)) * Nc + (j - 1)];
  if (i == j) eav = SELF_EDGE;
  if (eav == -1) eav = NO_EDGE;
  if (dtv != 1) eav = NO_EDGE;
  if (dtv == TASK_DIST) eav = TASK_EDGE;
  DT[idx] = dtv;
  EA[idx] = eav;
}

// Combined per-head f16 tables:
//  qkhe[h][T][c] (320x128): T<288: row = concat(qhop[T,h], khop[T,h]) (T<259)
//                           T>=288: row = concat(qedge[T-288,h], kedge[T-288,h])
//  vheT[h][d][T] (64x320):  T<288: vhop[T,h*64+d] ; T>=288: vedge[T-288,...]
__global__ __launch_bounds__(256) void tables_kernel(
    const float* __restrict__ qhop, const float* __restrict__ khop,
    const float* __restrict__ qedge, const float* __restrict__ kedge,
    const float* __restrict__ vhop, const float* __restrict__ vedge,
    _Float16* __restrict__ qkhe, _Float16* __restrict__ vheT) {
  int idx = blockIdx.x * 256 + threadIdx.x;
  const int n0 = Hc * TBP * 128;
  const int n1 = Hc * DKc * TBP;
  if (idx < n0) {
    int c = idx & 127;
    int T = (idx >> 7) % TBP;
    int h = idx / (TBP * 128);
    float v = 0.f;
    if (T < THOPc) {
      v = (c < 64) ? qhop[(size_t)T * Dc + h * DKc + c]
                   : khop[(size_t)T * Dc + h * DKc + (c - 64)];
    } else if (T >= EOFF && (T - EOFF) < TEDGEc) {
      int E = T - EOFF;
      v = (c < 64) ? qedge[(size_t)E * Dc + h * DKc + c]
                   : kedge[(size_t)E * Dc + h * DKc + (c - 64)];
    }
    qkhe[idx] = (_Float16)v;
    return;
  }
  idx -= n0;
  if (idx < n1) {
    int T = idx % TBP;
    int d = (idx / TBP) % DKc;
    int h = idx / (TBP * DKc);
    float v = 0.f;
    if (T < THOPc) v = vhop[(size_t)T * Dc + h * DKc + d];
    else if (T >= EOFF && (T - EOFF) < TEDGEc)
      v = vedge[(size_t)(T - EOFF) * Dc + h * DKc + d];
    vheT[idx] = (_Float16)v;
  }
}

// ---------------- LayerNorm: f32 in -> f16 out ----------------
__global__ __launch_bounds__(256) void ln_kernel(
    const float* __restrict__ x, const float* __restrict__ g,
    const float* __restrict__ bta, _Float16* __restrict__ y) {
  int row = blockIdx.x;
  const float* xr = x + (size_t)row * Dc;
  int tid = threadIdx.x, lane = tid & 31, w = tid >> 5;
  float s = 0.f, s2 = 0.f;
  for (int i = tid; i < Dc; i += 256) { float v = xr[i]; s += v; s2 += v * v; }
#pragma unroll
  for (int off = 16; off; off >>= 1) {
    s += __shfl_xor(s, off, 32);
    s2 += __shfl_xor(s2, off, 32);
  }
  __shared__ float rs[8], rs2[8];
  if (lane == 0) { rs[w] = s; rs2[w] = s2; }
  __syncthreads();
  float ts = 0.f, ts2 = 0.f;
#pragma unroll
  for (int j = 0; j < 8; ++j) { ts += rs[j]; ts2 += rs2[j]; }
  float m = ts / (float)Dc;
  float inv = rsqrtf(ts2 / (float)Dc - m * m + 1e-5f);
  for (int i = tid; i < Dc; i += 256)
    y[(size_t)row * Dc + i] = (_Float16)((xr[i] - m) * inv * g[i] + bta[i]);
}

// ---------------- Generic WMMA GEMM ----------------
// C[M,N] = act(A_f16[M,K] @ B_f32[K,N] + bias) (+ residual_f32); out f32/f16.
// 128x64 block tile, BK=32, 256 threads = 8 waves; each wave owns a 32x32
// macro tile -> 4 accumulators, 4 WMMAs per k-step.
template <int GELU>
__global__ __launch_bounds__(256) void gemm_kernel(
    const _Float16* __restrict__ A, const float* __restrict__ B,
    const float* __restrict__ bias, const float* __restrict__ residual,
    float* __restrict__ outF, _Float16* __restrict__ outH, int M, int N, int K) {
  __shared__ __align__(16) _Float16 sA[128][40];
  __shared__ __align__(16) _Float16 sB[64][40];  // transposed: [n][k]
  int tid = threadIdx.x, lane = tid & 31;
  int w = __builtin_amdgcn_readfirstlane(tid >> 5);
  int n0 = blockIdx.x * 64, m0 = blockIdx.y * 128;
  int mt0 = (w & 3) * 2, nt0 = (w >> 2) * 2;
  v8f acc00 = {}, acc01 = {}, acc10 = {}, acc11 = {};
  for (int k0 = 0; k0 < K; k0 += 32) {
#pragma unroll
    for (int j = 0; j < 2; ++j) {  // stage A tile: 2x16B chunks per thread
      int chunk = tid * 2 + j;
      int r = chunk >> 2, c = (chunk & 3) * 8;
      *reinterpret_cast<h8*>(&sA[r][c]) =
          *reinterpret_cast<const h8*>(A + (size_t)(m0 + r) * K + k0 + c);
    }
    for (int i = tid; i < 32 * 64; i += 256) {  // stage B tile, f32->f16, transpose
      int kk = i >> 6, nn = i & 63;
      sB[nn][kk] = (_Float16)B[(size_t)(k0 + kk) * N + n0 + nn];
    }
    if (k0 + 32 < K) {  // global_prefetch_b8 of next tiles
      __builtin_prefetch(A + (size_t)(m0 + (tid >> 1)) * K + k0 + 32, 0, 1);
      __builtin_prefetch(
          B + (size_t)(k0 + 32 + (tid >> 3)) * N + n0 + (tid & 7) * 8, 0, 1);
    }
    __syncthreads();
    v16h a0 = frag_load(&sA[mt0 * 16][0], 40);
    v16h a1 = frag_load(&sA[(mt0 + 1) * 16][0], 40);
    v16h b0 = frag_load(&sB[nt0 * 16][0], 40);
    v16h b1 = frag_load(&sB[(nt0 + 1) * 16][0], 40);
    acc00 = wmma32(a0, b0, acc00);
    acc01 = wmma32(a0, b1, acc01);
    acc10 = wmma32(a1, b0, acc10);
    acc11 = wmma32(a1, b1, acc11);
    __syncthreads();
  }
  int col = lane & 15, rb = (lane >> 4) * 8;
  auto epi = [&](v8f acc, int mt, int nt) {
    int nn = n0 + nt * 16 + col;
    float bv = bias ? bias[nn] : 0.f;
#pragma unroll
    for (int r = 0; r < 8; ++r) {
      int mm = m0 + mt * 16 + r + rb;
      float v = acc[r] + bv;
      if (GELU) v = 0.5f * v * (1.f + erff(v * 0.70710678118654752f));
      if (residual) v += residual[(size_t)mm * N + nn];
      if (outF) outF[(size_t)mm * N + nn] = v;
      if (outH) outH[(size_t)mm * N + nn] = (_Float16)v;
    }
  };
  epi(acc00, mt0, nt0);
  epi(acc01, mt0, nt0 + 1);
  epi(acc10, mt0 + 1, nt0);
  epi(acc11, mt0 + 1, nt0 + 1);
}

// ---------------- Fused relational attention ----------------
// One block = one (b, h, 16-row slice of s).  Branch-free WMMA pipeline with
// combined hop|edge bias bins:
//  biases (20 WMMA tiles) -> scores (2/tile) -> softmax -> ds_add_f32 segment
//  scatter -> output o = att@V (8) + sV@vheT (10).
__global__ __launch_bounds__(256) void attn_kernel(
    const _Float16* __restrict__ Q, const _Float16* __restrict__ K,
    const _Float16* __restrict__ V, const int* __restrict__ DT,
    const int* __restrict__ EA, const uint8_t* __restrict__ Mask,
    const _Float16* __restrict__ qkhe, const _Float16* __restrict__ vheT,
    _Float16* __restrict__ O) {
  __shared__ __align__(16) _Float16 sK[256][72];    // k rows (B-frag, scores)
  __shared__ __align__(16) _Float16 sVT[64][264];   // v transposed [d][t]
  __shared__ __align__(16) _Float16 sQK[16][136];   // concat(q,k) slice rows
  __shared__ __align__(16) float    sBias[16][TBP]; // hop|edge bias bins
  __shared__ __align__(16) float    sS[16][256];    // scores / att probs
  __shared__ __align__(16) _Float16 sAtt[16][264];  // att f16 (A-frag)
  __shared__ __align__(16) float    sV[16][TBP];    // segment sums hop|edge
  __shared__ __align__(16) _Float16 sVH[16][TBP];

  int tid = threadIdx.x, lane = tid & 31;
  int w = __builtin_amdgcn_readfirstlane(tid >> 5);
  int s0 = blockIdx.x * 16, h = blockIdx.y, b = blockIdx.z;
  const _Float16* Qg = Q + (size_t)b * Sc * Dc + h * DKc;
  const _Float16* Kg = K + (size_t)b * Sc * Dc + h * DKc;
  const _Float16* Vg = V + (size_t)b * Sc * Dc + h * DKc;
  const int* DTb = DT + (size_t)b * Sc * Sc;
  const int* EAb = EA + (size_t)b * Sc * Sc;
  const uint8_t* Mb = Mask + (size_t)b * Nc;

  for (int i = tid; i < 256 * 64; i += 256) {
    int t = i >> 6, d = i & 63;
    sK[t][d] = Kg[(size_t)t * Dc + d];
  }
  for (int i = tid; i < 64 * 256; i += 256) {
    int d = i >> 8, t = i & 255;
    sVT[d][t] = Vg[(size_t)t * Dc + d];
  }
  for (int i = tid; i < 16 * 128; i += 256) {
    int r = i >> 7, c = i & 127;
    sQK[r][c] = (c < 64) ? Qg[(size_t)(s0 + r) * Dc + c]
                         : Kg[(size_t)(s0 + r) * Dc + (c - 64)];
  }
  for (int i = tid; i < 16 * TBP; i += 256) sV[i / TBP][i % TBP] = 0.f;
  __syncthreads();

  // ---- bias GEMM: sBias[16,320] = cat(q,k) @ qkhe[h]^T (uniform, no branch)
  const _Float16* QKHE = qkhe + (size_t)h * TBP * 128;
  for (int tile = w; tile < TBP / 16; tile += 8) {
    const _Float16* Bt = QKHE + (size_t)tile * 16 * 128;
    v8f acc = {};
#pragma unroll
    for (int ks = 0; ks < 4; ++ks) {
      v16h a = frag_load(&sQK[0][0] + ks * 32, 136);
      v16h bb = frag_load(Bt + ks * 32, 128);
      acc = wmma32(a, bb, acc);
    }
    int col = lane & 15, rb = (lane >> 4) * 8;
#pragma unroll
    for (int r = 0; r < 8; ++r) sBias[r + rb][tile * 16 + col] = acc[r];
  }
  __syncthreads();

  // ---- scores: q@k^T + gathered biases, scale, mask ----
  for (int tile = w; tile < 16; tile += 8) {
    v8f acc = {};
#pragma unroll
    for (int ks = 0; ks < 2; ++ks) {
      v16h a = frag_load(&sQK[0][0] + ks * 32, 136);  // q half of cat
      v16h bb = frag_load(&sK[tile * 16][0] + ks * 32, 72);
      acc = wmma32(a, bb, acc);
    }
    int col = tile * 16 + (lane & 15);
    int rb = (lane >> 4) * 8;
    bool masked = (col > 0) && (Mb[col - 1] != 0);
#pragma unroll
    for (int r = 0; r < 8; ++r) {
      int row = r + rb;
      size_t pij = (size_t)(s0 + row) * Sc + col;
      float v = (acc[r] + sBias[row][DTb[pij]] + sBias[row][EOFF + EAb[pij]]) *
                SCALEc;
      sS[row][col] = masked ? -__builtin_inff() : v;
    }
  }
  __syncthreads();

  // ---- softmax: wave handles 2 rows ----
  for (int row = w * 2; row < w * 2 + 2; ++row) {
    float mx = -__builtin_inff();
    for (int t = lane; t < 256; t += 32) mx = fmaxf(mx, sS[row][t]);
#pragma unroll
    for (int off = 16; off; off >>= 1) mx = fmaxf(mx, __shfl_xor(mx, off, 32));
    float sum = 0.f;
    for (int t = lane; t < 256; t += 32) {
      float e = __expf(sS[row][t] - mx);
      sS[row][t] = e;
      sum += e;
    }
#pragma unroll
    for (int off = 16; off; off >>= 1) sum += __shfl_xor(sum, off, 32);
    float inv = 1.f / sum;
    for (int t = lane; t < 256; t += 32) {
      float a = sS[row][t] * inv;
      sS[row][t] = a;
      sAtt[row][t] = (_Float16)a;
    }
  }
  __syncthreads();

  // ---- segment sums into combined hop|edge bins (ds_add_f32 scatter) ----
  for (int i = tid; i < 16 * 256; i += 256) {
    int row = i >> 8, t = i & 255;
    float a = sS[row][t];
    size_t pij = (size_t)(s0 + row) * Sc + t;
    atomicAdd(&sV[row][DTb[pij]], a);
    atomicAdd(&sV[row][EOFF + EAb[pij]], a);
  }
  __syncthreads();
  for (int i = tid; i < 16 * TBP; i += 256) {
    int r = i / TBP, c = i % TBP;
    sVH[r][c] = (_Float16)sV[r][c];
  }
  __syncthreads();

  // ---- output: o = att@V + sV@vheT ; 4 col tiles, waves 0..3 ----
  if (w < 4) {
    v8f acc = {};
#pragma unroll
    for (int ks = 0; ks < 8; ++ks) {
      v16h a = frag_load(&sAtt[0][0] + ks * 32, 264);
      v16h bb = frag_load(&sVT[w * 16][0] + ks * 32, 264);
      acc = wmma32(a, bb, acc);
    }
    const _Float16* VHE = vheT + (size_t)h * DKc * TBP + (size_t)w * 16 * TBP;
#pragma unroll
    for (int ks = 0; ks < TBP / 32; ++ks) {
      v16h a = frag_load(&sVH[0][0] + ks * 32, TBP);
      v16h bb = frag_load(VHE + ks * 32, TBP);
      acc = wmma32(a, bb, acc);
    }
    _Float16* Og = O + (size_t)b * Sc * Dc + h * DKc + w * 16;
    int col = lane & 15, rb = (lane >> 4) * 8;
#pragma unroll
    for (int r = 0; r < 8; ++r)
      Og[(size_t)(s0 + r + rb) * Dc + col] = (_Float16)acc[r];
  }
}

// ---------------- final LN + linear head ----------------
__global__ __launch_bounds__(256) void final_kernel(
    const float* __restrict__ xt, const float* __restrict__ g,
    const float* __restrict__ bta, const float* __restrict__ Wout,
    const float* __restrict__ bout, float* __restrict__ out) {
  int bI = blockIdx.x;
  const float* xr = xt + (size_t)bI * Sc * Dc;  // s = 0 row
  int tid = threadIdx.x, lane = tid & 31, w = tid >> 5;
  float s = 0.f, s2 = 0.f;
  for (int i = tid; i < Dc; i += 256) { float v = xr[i]; s += v; s2 += v * v; }
#pragma unroll
  for (int off = 16; off; off >>= 1) {
    s += __shfl_xor(s, off, 32);
    s2 += __shfl_xor(s2, off, 32);
  }
  __shared__ float rs[8], rs2[8], rd[8];
  if (lane == 0) { rs[w] = s; rs2[w] = s2; }
  __syncthreads();
  float ts = 0.f, ts2 = 0.f;
#pragma unroll
  for (int j = 0; j < 8; ++j) { ts += rs[j]; ts2 += rs2[j]; }
  float m = ts / (float)Dc;
  float inv = rsqrtf(ts2 / (float)Dc - m * m + 1e-5f);
  float dot = 0.f;
  for (int i = tid; i < Dc; i += 256)
    dot += ((xr[i] - m) * inv * g[i] + bta[i]) * Wout[i];
#pragma unroll
  for (int off = 16; off; off >>= 1) dot += __shfl_xor(dot, off, 32);
  if (lane == 0) rd[w] = dot;
  __syncthreads();
  if (tid == 0) {
    float td = 0.f;
#pragma unroll
    for (int j = 0; j < 8; ++j) td += rd[j];
    out[bI] = td + bout[0];
  }
}

// ---------------- host ----------------
extern "C" void kernel_launch(void* const* d_in, const int* in_sizes, int n_in,
                              void* d_out, int out_size, void* d_ws,
                              size_t ws_size, hipStream_t stream) {
  (void)in_sizes; (void)n_in; (void)out_size; (void)ws_size;
  const int*     node_x    = (const int*)d_in[0];
  const uint8_t* mask      = (const uint8_t*)d_in[1];
  const int*     distance  = (const int*)d_in[2];
  const int*     edge_attr = (const int*)d_in[3];
  const float*   node_emb  = (const float*)d_in[4];
  const float*   task_tok  = (const float*)d_in[5];
  const float* qhop  = (const float*)d_in[6];
  const float* qedge = (const float*)d_in[7];
  const float* khop  = (const float*)d_in[8];
  const float* kedge = (const float*)d_in[9];
  const float* vhop  = (const float*)d_in[10];
  const float* vedge = (const float*)d_in[11];
  const float* ln1_g = (const float*)d_in[12];
  const float* ln1_b = (const float*)d_in[13];
  const float* Wq = (const float*)d_in[14];
  const float* bq = (const float*)d_in[15];
  const float* Wk = (const float*)d_in[16];
  const float* bk = (const float*)d_in[17];
  const float* Wv = (const float*)d_in[18];
  const float* bv = (const float*)d_in[19];
  const float* Wo = (const float*)d_in[20];
  const float* bo = (const float*)d_in[21];
  const float* ln2_g = (const float*)d_in[22];
  const float* ln2_b = (const float*)d_in[23];
  const float* W1 = (const float*)d_in[24];
  const float* b1 = (const float*)d_in[25];
  const float* W2 = (const float*)d_in[26];
  const float* b2 = (const float*)d_in[27];
  const float* fln_g = (const float*)d_in[28];
  const float* fln_b = (const float*)d_in[29];
  const float* Wout  = (const float*)d_in[30];
  const float* bout  = (const float*)d_in[31];

  char* ws = (char*)d_ws;
  size_t off = 0;
  auto alloc = [&](size_t bytes) -> void* {
    void* p = ws + off;
    off = (off + bytes + 255) & ~(size_t)255;
    return p;
  };
  const size_t ROWS = (size_t)Bc * Sc;  // 4096
  float*    xt  = (float*)alloc(ROWS * Dc * 4);
  _Float16* yh  = (_Float16*)alloc(ROWS * Dc * 2);
  _Float16* qf  = (_Float16*)alloc(ROWS * Dc * 2);
  _Float16* kf  = (_Float16*)alloc(ROWS * Dc * 2);
  _Float16* vf  = (_Float16*)alloc(ROWS * Dc * 2);
  _Float16* of  = (_Float16*)alloc(ROWS * Dc * 2);
  _Float16* h1  = (_Float16*)alloc(ROWS * DFFc * 2);
  int* DT = (int*)alloc((size_t)Bc * Sc * Sc * 4);
  int* EA = (int*)alloc((size_t)Bc * Sc * Sc * 4);
  _Float16* qkhe = (_Float16*)alloc((size_t)Hc * TBP * 128 * 2);
  _Float16* vheT = (_Float16*)alloc((size_t)Hc * DKc * TBP * 2);

  // prep
  embed_kernel<<<(Bc * Sc * Dc + 255) / 256, 256, 0, stream>>>(node_x, node_emb,
                                                               task_tok, xt);
  dist_edge_kernel<<<(Bc * Sc * Sc + 255) / 256, 256, 0, stream>>>(
      distance, edge_attr, DT, EA);
  {
    int total = Hc * TBP * 128 + Hc * DKc * TBP;
    tables_kernel<<<(total + 255) / 256, 256, 0, stream>>>(
        qhop, khop, qedge, kedge, vhop, vedge, qkhe, vheT);
  }

  dim3 gProj(Dc / 64, ROWS / 128);    // (12, 32)
  dim3 gFF1(DFFc / 64, ROWS / 128);   // (48, 32)
  dim3 gAttn(Sc / 16, Hc, Bc);        // (16, 12, 16)

  for (int i = 0; i < Lc; ++i) {
    ln_kernel<<<ROWS, 256, 0, stream>>>(xt, ln1_g + (size_t)i * Dc,
                                        ln1_b + (size_t)i * Dc, yh);
    gemm_kernel<0><<<gProj, 256, 0, stream>>>(
        yh, Wq + (size_t)i * Dc * Dc, bq + (size_t)i * Dc, nullptr, nullptr, qf,
        (int)ROWS, Dc, Dc);
    gemm_kernel<0><<<gProj, 256, 0, stream>>>(
        yh, Wk + (size_t)i * Dc * Dc, bk + (size_t)i * Dc, nullptr, nullptr, kf,
        (int)ROWS, Dc, Dc);
    gemm_kernel<0><<<gProj, 256, 0, stream>>>(
        yh, Wv + (size_t)i * Dc * Dc, bv + (size_t)i * Dc, nullptr, nullptr, vf,
        (int)ROWS, Dc, Dc);
    attn_kernel<<<gAttn, 256, 0, stream>>>(qf, kf, vf, DT, EA, mask, qkhe,
                                           vheT, of);
    gemm_kernel<0><<<gProj, 256, 0, stream>>>(
        of, Wo + (size_t)i * Dc * Dc, bo + (size_t)i * Dc, xt, xt, nullptr,
        (int)ROWS, Dc, Dc);
    ln_kernel<<<ROWS, 256, 0, stream>>>(xt, ln2_g + (size_t)i * Dc,
                                        ln2_b + (size_t)i * Dc, yh);
    gemm_kernel<1><<<gFF1, 256, 0, stream>>>(
        yh, W1 + (size_t)i * Dc * DFFc, b1 + (size_t)i * DFFc, nullptr, nullptr,
        h1, (int)ROWS, DFFc, Dc);
    gemm_kernel<0><<<gProj, 256, 0, stream>>>(
        h1, W2 + (size_t)i * DFFc * Dc, b2 + (size_t)i * Dc, xt, xt, nullptr,
        (int)ROWS, Dc, DFFc);
  }

  final_kernel<<<Bc, 256, 0, stream>>>(xt, fln_g, fln_b, Wout, bout,
                                       (float*)d_out);
}
